// ScaledDotProductAttention_6923487281202
// MI455X (gfx1250) — compile-verified, matrix operand order checked by reading, not run on hardware
//
#include <hip/hip_runtime.h>

typedef __attribute__((ext_vector_type(16))) _Float16 v16h;
typedef __attribute__((ext_vector_type(8)))  float    v8f;

#define L_SEQ   2048
#define HDIM    64
#define QTILE   16      // q rows per wave
#define NWAVES  4       // waves per block -> 64 q rows per block
#define KBLK    32      // keys per main-loop iteration

#define SB_STRIDE 34    // f32 stride for S tile rows (bank-conflict pad)
#define VT_STRIDE 36    // f16 stride for transposed-V rows (bank-conflict pad)

// ---- Tensor Data Mover support (clang-23 / therock 6-arg builtin form) ----
#if defined(__has_builtin)
#  if __has_builtin(__builtin_amdgcn_tensor_load_to_lds) && __has_builtin(__builtin_amdgcn_s_wait_tensorcnt)
#    define USE_TDM 1
#  endif
#endif
#ifndef USE_TDM
#  define USE_TDM 0
#endif

#define MROW_STRIDE 36                       // 32B mask row + 4B TDM pad (bank-conflict free)
#define MBUF_BYTES  (QTILE * NWAVES * MROW_STRIDE)

#if USE_TDM
typedef __attribute__((ext_vector_type(4))) unsigned int u32x4;
typedef __attribute__((ext_vector_type(8))) int          i32x8;
typedef __attribute__((ext_vector_type(4))) int          i32x4;

// Issue one TDM transfer: 64-row x 32-byte mask tile (row pitch 2048B) -> LDS,
// with 1 DWORD of padding inserted every 8 DWORDs (LDS row stride = 36B).
__device__ __forceinline__ void tdm_load_mask(unsigned lds_off, unsigned long long gaddr)
{
    u32x4 g0;
    g0[0] = 1u;                                                    // count=1 (valid user D#)
    g0[1] = lds_off;                                               // lds_addr
    g0[2] = (unsigned)(gaddr & 0xffffffffu);                       // global_addr[31:0]
    g0[3] = (unsigned)((gaddr >> 32) & 0x01ffffffu) | (2u << 30);  // global_addr[56:32] | type=2
    i32x8 g1;
    g1[0] = (1 << 20) | (2 << 22);                 // pad_enable, pad_interval=8 DW, pad_amount=1 DW
    g1[1] = (int)((unsigned)(L_SEQ & 0xffffu) << 16);   // tensor_dim0 = 2048 elements (1B each)
    g1[2] = (int)((unsigned)(QTILE * NWAVES) << 16);    // tensor_dim1 = 64 rows
    g1[3] = (int)(32u << 16);                           // tile_dim0 = 32
    g1[4] = (int)(QTILE * NWAVES);                      // tile_dim1 = 64
    g1[5] = L_SEQ;                                      // tensor_dim0_stride = 2048
    g1[6] = 0;
    g1[7] = 0;
    i32x4 gz4 = {0, 0, 0, 0};                           // groups 2/3 unused (2D tile)
    i32x8 gz8 = {0, 0, 0, 0, 0, 0, 0, 0};               // trailing group (6-arg form)
    __builtin_amdgcn_tensor_load_to_lds(g0, g1, gz4, gz4, gz8, 0);
}
#endif

__global__ __launch_bounds__(128)
void attn_fused_kernel(const float* __restrict__ q,
                       const float* __restrict__ k,
                       const float* __restrict__ v,
                       const unsigned char* __restrict__ mask,
                       float* __restrict__ out)
{
    __shared__ float    sS[NWAVES * QTILE * SB_STRIDE];   // raw scaled scores, per-wave region
    __shared__ _Float16 sVt[HDIM * VT_STRIDE];            // V block transposed [d][key], f16
    __shared__ float    sRS[NWAVES * QTILE];              // row sums
#if USE_TDM
    __shared__ __align__(16) unsigned char sMask[2][MBUF_BYTES];  // double-buffered mask tiles
#endif

    const int tid  = threadIdx.x;
    const int wave = tid >> 5;
    const int lane = tid & 31;
    const int hf   = lane >> 4;      // half-wave id (0/1)
    const int l16  = lane & 15;

    const int b    = blockIdx.x >> 5;        // 32 q-blocks per batch
    const int qblk = blockIdx.x & 31;
    const int q0   = qblk * (QTILE * NWAVES) + wave * QTILE;

    // A/B operand per-lane K split (ISA 7.12.2, 16-bit 16x32 layout)
    const int koff0 = hf * 8;        // elements 0..7  <- K = koff0..koff0+7
    const int koff1 = 16 + hf * 8;   // elements 8..15 <- K = koff1..koff1+7

    // ---- Q tile -> f16 A operands (K = 0..31 and 32..63), loaded once ----
    v16h Aq0, Aq1;
    {
        const float* qrow = q + (size_t)(b * L_SEQ + q0 + l16) * HDIM;
        #pragma unroll
        for (int e = 0; e < 8; ++e) {
            Aq0[e]     = (_Float16)qrow[koff0 + e];
            Aq0[e + 8] = (_Float16)qrow[koff1 + e];
            Aq1[e]     = (_Float16)qrow[32 + koff0 + e];
            Aq1[e + 8] = (_Float16)qrow[32 + koff1 + e];
        }
    }

    v8f accO[4];
    #pragma unroll
    for (int t = 0; t < 4; ++t) accO[t] = (v8f){0.f,0.f,0.f,0.f,0.f,0.f,0.f,0.f};
    float rs = 0.f;   // per-lane partial row sum (row = l16)

    const float* kbase = k + (size_t)b * L_SEQ * HDIM;
    const float* vbase = v + (size_t)b * L_SEQ * HDIM;

#if USE_TDM
    const unsigned long long mgbase =
        (unsigned long long)(size_t)(mask + ((size_t)b * L_SEQ +
                                             (size_t)(qblk * QTILE * NWAVES)) * L_SEQ);
    const unsigned mlds = (unsigned)(size_t)(void*)&sMask[0][0];
    if (wave == 0) tdm_load_mask(mlds, mgbase);            // pipeline preamble: block 0
#else
    const unsigned char* mrow = mask + ((size_t)b * L_SEQ + (size_t)(q0 + l16)) * L_SEQ;
#endif

    float* sSw = sS + wave * (QTILE * SB_STRIDE);

    for (int j = 0; j < L_SEQ; j += KBLK) {
        // ---- cooperative stage: V block (32 keys x 64 d) transposed to LDS as f16 ----
        __syncthreads();   // all waves done reading sVt / sMask of previous block
        {
            const int key   = tid >> 2;          // 0..31
            const int dpart = (tid & 3) * 16;    // 0,16,32,48
            const float* vp = vbase + (size_t)(j + key) * HDIM + dpart;
            #pragma unroll
            for (int e = 0; e < 16; ++e)
                sVt[(dpart + e) * VT_STRIDE + key] = (_Float16)vp[e];
        }
        if (j + KBLK < L_SEQ) {
            // prefetch next K/V block -> global_prefetch_b8
            __builtin_prefetch(kbase + (size_t)(j + KBLK + l16) * HDIM, 0, 1);
            __builtin_prefetch(vbase + (size_t)(j + KBLK + l16) * HDIM, 0, 1);
        }
#if USE_TDM
        if (wave == 0) {
            if (j + KBLK < L_SEQ) {
                // issue next block's mask DMA into the other buffer, then make sure
                // the CURRENT block's DMA (issued last iteration / preamble) is done
                tdm_load_mask(mlds + (unsigned)((((j >> 5) + 1) & 1) * MBUF_BYTES),
                              mgbase + (unsigned long long)(j + KBLK));
                __builtin_amdgcn_s_wait_tensorcnt(1);
            } else {
                __builtin_amdgcn_s_wait_tensorcnt(0);
            }
        }
#endif
        __syncthreads();   // sVt + mask tile ready

        // ---- S = (Q K^T) * scale for two 16-key chunks ----
        #pragma unroll
        for (int c = 0; c < 2; ++c) {
            const float* kp = kbase + (size_t)(j + c * 16 + l16) * HDIM;
            v16h Bk0, Bk1;   // B operand: lane = key column, elements over d
            #pragma unroll
            for (int e = 0; e < 8; ++e) {
                Bk0[e]     = (_Float16)kp[koff0 + e];
                Bk0[e + 8] = (_Float16)kp[koff1 + e];
                Bk1[e]     = (_Float16)kp[32 + koff0 + e];
                Bk1[e + 8] = (_Float16)kp[32 + koff1 + e];
            }
            v8f s = (v8f){0.f,0.f,0.f,0.f,0.f,0.f,0.f,0.f};
            s = __builtin_amdgcn_wmma_f32_16x16x32_f16(false, Aq0, false, Bk0,
                                                       (short)0, s, false, false);
            s = __builtin_amdgcn_wmma_f32_16x16x32_f16(false, Aq1, false, Bk1,
                                                       (short)0, s, false, false);
            // C layout: VGPR r, lane -> row (r + 8*hf), col l16. Store scaled scores.
            #pragma unroll
            for (int r = 0; r < 8; ++r)
                sSw[(r + 8 * hf) * SB_STRIDE + c * 16 + l16] = s[r] * 0.125f;
        }

        // ---- relayout to A format: mask -> exp -> rowsum -> f16 P tile ----
        const float* sr = sSw + l16 * SB_STRIDE;   // this lane's row
#if USE_TDM
        const unsigned char* mp =
            &sMask[(j >> 5) & 1][(wave * QTILE + l16) * MROW_STRIDE];
#else
        const unsigned char* mp = mrow + j;
#endif
        v16h Ap;
        #pragma unroll
        for (int e = 0; e < 8; ++e) {
            const float s0 = mp[koff0 + e] ? 0.f : sr[koff0 + e];
            const float s1 = mp[koff1 + e] ? 0.f : sr[koff1 + e];
            const float e0 = __expf(s0);           // masked -> exp(0)=1, matches ref
            const float e1 = __expf(s1);
            rs += e0 + e1;
            Ap[e]     = (_Float16)e0;
            Ap[e + 8] = (_Float16)e1;
        }

        // ---- O += P x V  (B operand from transposed V in LDS: contiguous) ----
        #pragma unroll
        for (int t = 0; t < 4; ++t) {
            const _Float16* vt = sVt + (t * 16 + l16) * VT_STRIDE;  // lane = d column
            v16h Bv;
            #pragma unroll
            for (int e = 0; e < 8; ++e) {
                Bv[e]     = vt[koff0 + e];   // K = key offset in block
                Bv[e + 8] = vt[koff1 + e];
            }
            accO[t] = __builtin_amdgcn_wmma_f32_16x16x32_f16(false, Ap, false, Bv,
                                                             (short)0, accO[t], false, false);
        }
    }

    // ---- epilogue: finish row sums, normalize, store ----
    rs += __shfl_xor(rs, 16, 32);        // lane l and l^16 hold the same row
    if (hf == 0) sRS[wave * QTILE + l16] = rs;   // same-wave LDS: in-order, no barrier needed

    #pragma unroll
    for (int r = 0; r < 8; ++r) {
        const int row = r + 8 * hf;
        const float inv = 1.f / sRS[wave * QTILE + row];
        const size_t obase = (size_t)(b * L_SEQ + q0 + row) * HDIM;
        #pragma unroll
        for (int t = 0; t < 4; ++t)
            out[obase + t * 16 + l16] = accO[t][r] * inv;
    }
}

extern "C" void kernel_launch(void* const* d_in, const int* in_sizes, int n_in,
                              void* d_out, int out_size, void* d_ws, size_t ws_size,
                              hipStream_t stream) {
    (void)n_in; (void)out_size; (void)d_ws; (void)ws_size;
    const float* q = (const float*)d_in[0];
    const float* k = (const float*)d_in[1];
    const float* v = (const float*)d_in[2];
    const unsigned char* mask = (const unsigned char*)d_in[3];  // numpy bool -> 1 byte
    float* out = (float*)d_out;

    const int B = in_sizes[0] / (L_SEQ * HDIM);                 // = 16
    dim3 grid(B * (L_SEQ / (QTILE * NWAVES)));                  // 16 * 32 = 512 blocks
    dim3 block(NWAVES * 32);                                    // 4 waves (wave32)
    attn_fused_kernel<<<grid, block, 0, stream>>>(q, k, v, mask, out);
}
